// DGMmodule_58308476011161
// MI455X (gfx1250) — matrix operand order repaired and checked by reading, not compiled, not run on hardware
//
#include <hip/hip_runtime.h>
#include <math.h>

// ---------------------------------------------------------------------------
// Types for CDNA5 WMMA (wave32): v_wmma_f32_16x16x32_bf16
// ---------------------------------------------------------------------------
typedef __bf16 bf16_t;
typedef bf16_t bf16x4  __attribute__((ext_vector_type(4)));
typedef bf16_t bf16x8  __attribute__((ext_vector_type(8)));
typedef bf16_t bf16x16 __attribute__((ext_vector_type(16)));
typedef float  f32x8   __attribute__((ext_vector_type(8)));

#define LDS_STRIDE 40   // bf16 units per row: 80 B, multiple of 16 B (b128-aligned)

// ---------------------------------------------------------------------------
// LDS fragment loaders per the CDNA5 ISA 16-bit operand layouts.
// A (16x32): lane m=L&15, half g=L>>4; elems 0..7  -> K = g*8 + 0..7
//                                      elems 8..15 -> K = 16 + g*8 + 0..7
// B (32x16, staged n-major): lane n=L&15, half g;  same K order as A
// ---------------------------------------------------------------------------
__device__ __forceinline__ bf16x16 load_frag_a(const bf16_t* lds, int row0, int lane) {
  int m = lane & 15, g = lane >> 4;
  const bf16_t* p = lds + (row0 + m) * LDS_STRIDE + g * 8;
  bf16x8 lo = *(const bf16x8*)(p);
  bf16x8 hi = *(const bf16x8*)(p + 16);
  return __builtin_shufflevector(lo, hi, 0,1,2,3,4,5,6,7,8,9,10,11,12,13,14,15);
}

__device__ __forceinline__ bf16x16 load_frag_b(const bf16_t* lds, int row0, int lane) {
  int n = lane & 15, g = lane >> 4;
  const bf16_t* p = lds + (row0 + n) * LDS_STRIDE + g * 8;
  bf16x8 lo = *(const bf16x8*)(p);
  bf16x8 hi = *(const bf16x8*)(p + 16);
  return __builtin_shufflevector(lo, hi, 0,1,2,3,4,5,6,7,8,9,10,11,12,13,14,15);
}

// ---------------------------------------------------------------------------
// Async staging: 128x32 bf16 tile (row-major, leading dim ld) -> LDS (stride 40)
// via global_load_async_to_lds_b128 (ASYNCcnt tracked; no VGPR round trip).
// 256 threads x 2 chunks of 8 bf16 (16 B) each. All addresses 16 B aligned.
// ---------------------------------------------------------------------------
__device__ __forceinline__ void stage_async(bf16_t* ldsbuf, const bf16_t* __restrict__ src,
                                            long ld, int k0, int tid) {
#pragma unroll
  for (int i = 0; i < 2; ++i) {
    int c = tid + i * 256;            // 0..511
    int row = c >> 2;                 // 0..127
    int col = (c & 3) * 8;            // 0,8,16,24
    unsigned lofs = (unsigned)(uintptr_t)(ldsbuf + row * LDS_STRIDE + col);
    const bf16_t* g = src + (long)row * ld + k0 + col;
    asm volatile("global_load_async_to_lds_b128 %0, %1, off"
                 :: "v"(lofs), "v"(g) : "memory");
  }
}

__device__ __forceinline__ void wait_async_all() {
  asm volatile("s_wait_asynccnt 0x0" ::: "memory");
}

// ---------------------------------------------------------------------------
// GEMM: C[M,N] = act(A[M,K] @ Bt[N,K]^T + bias); bf16 operands, f32 accum.
// Block tile 128x128, 8 waves (4x2), wave tile 32x64 (2x4 16x16 frags).
// Double-buffered LDS; async tile s+1 streams in under the WMMAs of tile s.
// M,N multiples of 128; K multiple of 32 (true at all call sites).
// Output: f32 (Cf) or bf16 (Cb) -- exactly one non-null.
// ---------------------------------------------------------------------------
__global__ __launch_bounds__(256) void gemm_bf16_wmma_kernel(
    const bf16_t* __restrict__ A, const bf16_t* __restrict__ Bt,
    const float* __restrict__ bias, float* __restrict__ Cf, bf16_t* __restrict__ Cb,
    int M, int N, int K, int leaky) {
  __shared__ bf16_t lA[2][128 * LDS_STRIDE];
  __shared__ bf16_t lB[2][128 * LDS_STRIDE];
  const int tid = threadIdx.x;
  const int lane = tid & 31, wave = tid >> 5;
  const int wm = wave >> 1, wn = wave & 1;          // 4x2 wave grid
  const long bm = (long)blockIdx.x * 128;
  const long bn = (long)blockIdx.y * 128;
  const bf16_t* Ab = A + bm * (long)K;
  const bf16_t* Bb = Bt + bn * (long)K;
  const int steps = K >> 5;

  stage_async(lA[0], Ab, K, 0, tid);
  stage_async(lB[0], Bb, K, 0, tid);
  wait_async_all();
  __syncthreads();

  f32x8 acc[2][4] = {};
  for (int s = 0; s < steps; ++s) {
    const int cur = s & 1;
    if (s + 1 < steps) {                            // stream next tile under WMMA
      stage_async(lA[cur ^ 1], Ab, K, (s + 1) * 32, tid);
      stage_async(lB[cur ^ 1], Bb, K, (s + 1) * 32, tid);
    }
    bf16x16 af[2], bfv[4];
#pragma unroll
    for (int mi = 0; mi < 2; ++mi) af[mi] = load_frag_a(lA[cur], wm * 32 + mi * 16, lane);
#pragma unroll
    for (int ni = 0; ni < 4; ++ni) bfv[ni] = load_frag_b(lB[cur], wn * 64 + ni * 16, lane);
#pragma unroll
    for (int mi = 0; mi < 2; ++mi)
#pragma unroll
      for (int ni = 0; ni < 4; ++ni)
        acc[mi][ni] = __builtin_amdgcn_wmma_f32_16x16x32_bf16(
            false, af[mi], false, bfv[ni], (short)0, acc[mi][ni], false, false);
    wait_async_all();
    __syncthreads();
  }

  const int nl = lane & 15, g = lane >> 4;
#pragma unroll
  for (int mi = 0; mi < 2; ++mi)
#pragma unroll
    for (int ni = 0; ni < 4; ++ni) {
      long col = bn + wn * 64 + ni * 16 + nl;
      float bv = bias ? bias[col] : 0.0f;
#pragma unroll
      for (int r = 0; r < 8; ++r) {
        long row = bm + wm * 32 + mi * 16 + (r + 8 * g);
        float v = acc[mi][ni][r] + bv;
        if (leaky) v = (v >= 0.0f) ? v : 0.1f * v;
        if (Cb) Cb[row * N + col] = (bf16_t)v;
        else    Cf[row * N + col] = v;
      }
    }
}

// ---------------------------------------------------------------------------
// out[i,j] = alpha * ( S[i] + S[j] - 2 * sum_b Y_b @ Y_b^T )  over n x n.
// Y: bf16, batches of (n x d) rows at batchStride (elements). alpha from tptr
// (exp(clip(t,-5,5))) or 1. Flattened (b,k0) pipeline, double-buffered async.
// ---------------------------------------------------------------------------
__global__ __launch_bounds__(256) void pairsq_wmma_kernel(
    const bf16_t* __restrict__ Y, const float* __restrict__ S,
    const float* __restrict__ tptr, float* __restrict__ out,
    int n, int d, int batches, long batchStride) {
  __shared__ bf16_t lA[2][128 * LDS_STRIDE];
  __shared__ bf16_t lB[2][128 * LDS_STRIDE];
  const int tid = threadIdx.x;
  const int lane = tid & 31, wave = tid >> 5;
  const int wm = wave >> 1, wn = wave & 1;
  const long bm = (long)blockIdx.x * 128;
  const long bn = (long)blockIdx.y * 128;
  const int kd = d >> 5;
  const int steps = batches * kd;

  stage_async(lA[0], Y + bm * (long)d, d, 0, tid);
  stage_async(lB[0], Y + bn * (long)d, d, 0, tid);
  wait_async_all();
  __syncthreads();

  f32x8 acc[2][4] = {};
  for (int s = 0; s < steps; ++s) {
    const int cur = s & 1;
    if (s + 1 < steps) {
      int b1 = (s + 1) / kd, k1 = ((s + 1) % kd) * 32;
      const bf16_t* Yb = Y + (long)b1 * batchStride;
      stage_async(lA[cur ^ 1], Yb + bm * (long)d, d, k1, tid);
      stage_async(lB[cur ^ 1], Yb + bn * (long)d, d, k1, tid);
    }
    bf16x16 af[2], bfv[4];
#pragma unroll
    for (int mi = 0; mi < 2; ++mi) af[mi] = load_frag_a(lA[cur], wm * 32 + mi * 16, lane);
#pragma unroll
    for (int ni = 0; ni < 4; ++ni) bfv[ni] = load_frag_b(lB[cur], wn * 64 + ni * 16, lane);
#pragma unroll
    for (int mi = 0; mi < 2; ++mi)
#pragma unroll
      for (int ni = 0; ni < 4; ++ni)
        acc[mi][ni] = __builtin_amdgcn_wmma_f32_16x16x32_bf16(
            false, af[mi], false, bfv[ni], (short)0, acc[mi][ni], false, false);
    wait_async_all();
    __syncthreads();
  }

  float alpha = 1.0f;
  if (tptr) alpha = __expf(fminf(fmaxf(tptr[0], -5.0f), 5.0f));
  const int nl = lane & 15, g = lane >> 4;
#pragma unroll
  for (int mi = 0; mi < 2; ++mi)
#pragma unroll
    for (int ni = 0; ni < 4; ++ni) {
      long col = bn + wn * 64 + ni * 16 + nl;
      float sj = S[col];
#pragma unroll
      for (int r = 0; r < 8; ++r) {
        long row = bm + wm * 32 + mi * 16 + (r + 8 * g);
        out[row * n + col] = alpha * (S[row] + sj - 2.0f * acc[mi][ni][r]);
      }
    }
}

// ---------------------------------------------------------------------------
// One-time conversion / transpose kernels (cheap, run once per launch)
// ---------------------------------------------------------------------------
__global__ __launch_bounds__(256) void convert_bf16_kernel(const float* __restrict__ in,
                                                           bf16_t* __restrict__ out,
                                                           long count4) {  // count/4
  long i = ((long)blockIdx.x * blockDim.x + threadIdx.x);
  if (i >= count4) return;
  float4 v = *(const float4*)(in + i * 4);
  bf16x4 w = { (bf16_t)v.x, (bf16_t)v.y, (bf16_t)v.z, (bf16_t)v.w };
  *(bf16x4*)(out + i * 4) = w;
}

// Wt[n*K + k] = bf16(W[k*N + n]); K,N multiples of 32.
__global__ __launch_bounds__(256) void transpose_bf16_kernel(const float* __restrict__ W,
                                                             bf16_t* __restrict__ Wt,
                                                             int K, int N) {
  __shared__ float t[32][33];
  int k0 = blockIdx.x * 32, n0 = blockIdx.y * 32;
  int tx = threadIdx.x & 31, ty = threadIdx.x >> 5;     // ty: 0..7
  for (int r = ty; r < 32; r += 8) t[r][tx] = W[(long)(k0 + r) * N + n0 + tx];
  __syncthreads();
  for (int r = ty; r < 32; r += 8) Wt[(long)(n0 + r) * K + k0 + tx] = (bf16_t)t[tx][r];
}

// ---------------------------------------------------------------------------
// Small helper kernels
// ---------------------------------------------------------------------------
__global__ __launch_bounds__(256) void rowsq_kernel(const float* __restrict__ in,
                                                    float* __restrict__ out, int cols) {
  long row = blockIdx.x;
  float s = 0.0f;
  for (int c = threadIdx.x; c < cols; c += 256) {
    float v = in[row * cols + c];
    s += v * v;
  }
  __shared__ float red[256];
  red[threadIdx.x] = s;
  __syncthreads();
  for (int off = 128; off > 0; off >>= 1) {
    if (threadIdx.x < off) red[threadIdx.x] += red[threadIdx.x + off];
    __syncthreads();
  }
  if (threadIdx.x == 0) out[row] = red[0];
}

__global__ void sumS_kernel(const float* __restrict__ sq2, float* __restrict__ S,
                            int n, int batches) {
  int i = blockIdx.x * blockDim.x + threadIdx.x;
  if (i >= n) return;
  float s = 0.0f;
  for (int b = 0; b < batches; ++b) s += sq2[(long)b * n + i];
  S[i] = s;
}

// Per-row top-k smallest of dlq (stable: smallest index on ties), n <= 768.
__global__ __launch_bounds__(256) void topk_kernel(const float* __restrict__ dlq,
                                                   int* __restrict__ idx, int n, int k) {
  __shared__ float vals[768];
  __shared__ float rv[256];
  __shared__ int ri[256];
  long row = blockIdx.x;
  for (int c = threadIdx.x; c < n; c += 256) vals[c] = dlq[row * n + c];
  __syncthreads();
  for (int t = 0; t < k; ++t) {
    float bv = INFINITY;
    int bi = n;
    for (int c = threadIdx.x; c < n; c += 256) {
      float v = vals[c];
      if (v < bv || (v == bv && c < bi)) { bv = v; bi = c; }
    }
    rv[threadIdx.x] = bv; ri[threadIdx.x] = bi;
    __syncthreads();
    for (int off = 128; off > 0; off >>= 1) {
      if (threadIdx.x < off) {
        float ov = rv[threadIdx.x + off]; int oi = ri[threadIdx.x + off];
        if (ov < rv[threadIdx.x] || (ov == rv[threadIdx.x] && oi < ri[threadIdx.x])) {
          rv[threadIdx.x] = ov; ri[threadIdx.x] = oi;
        }
      }
      __syncthreads();
    }
    if (threadIdx.x == 0) { idx[row * k + t] = ri[0]; vals[ri[0]] = INFINITY; }
    __syncthreads();
  }
}

__global__ void edges_kernel(const int* __restrict__ idx, float* __restrict__ eout,
                             int n, int k, int bp) {
  int e = blockIdx.x * blockDim.x + threadIdx.x;
  int total = bp * n * k;
  if (e >= total) return;
  int b = e / (n * k);
  int r = e % (n * k);
  int i = r / k;
  eout[e] = (float)(i + b * n);
  eout[total + e] = (float)(idx[r] + b * n);
}

__global__ __launch_bounds__(256) void logprobs_kernel(
    const float* __restrict__ xs, const float* __restrict__ sq2,
    const int* __restrict__ idx, const float* __restrict__ tptr,
    float* __restrict__ lp, int n, int d, int k) {
  int bi = blockIdx.x;            // 0 .. bp*n-1
  int b = bi / n, i = bi % n;
  int lane = threadIdx.x & 31;
  int kk = threadIdx.x >> 5;      // 8 waves -> 8 neighbors
  const float* xrow = xs + ((long)b * n + i) * d;
  int j = idx[(long)i * k + kk];
  const float* yrow = xs + (long)j * d;
  float s = 0.0f;
  for (int c = lane; c < d; c += 32) s += xrow[c] * yrow[c];
#pragma unroll
  for (int off = 16; off > 0; off >>= 1) s += __shfl_xor(s, off, 32);
  if (lane == 0) {
    float temp = __expf(fminf(fmaxf(tptr[0], -5.0f), 5.0f));
    lp[((long)b * n + i) * k + kk] = -(sq2[j] + sq2[(long)b * n + i] - 2.0f * s) * temp;
  }
}

// ---------------------------------------------------------------------------
// Host launcher
// ---------------------------------------------------------------------------
extern "C" void kernel_launch(void* const* d_in, const int* in_sizes, int n_in,
                              void* d_out, int out_size, void* d_ws, size_t ws_size,
                              hipStream_t stream) {
  (void)in_sizes; (void)n_in; (void)out_size; (void)ws_size;
  const float* x_pre = (const float*)d_in[0];   // (2,16,768,256)
  // d_in[1] = A, unused by the MLP branch
  const float* W1 = (const float*)d_in[2];      // (256,512)
  const float* b1 = (const float*)d_in[3];      // (512)
  const float* W2 = (const float*)d_in[4];      // (512,768)
  const float* b2 = (const float*)d_in[5];      // (768)
  const float* temperature = (const float*)d_in[6];

  const int N = 768, F = 256, Hd = 512, K = 8, BP = 16;
  const long M = 2L * 16 * N;                   // 24576 MLP rows
  const long NN = (long)N * N;

  // workspace layout (byte offsets; x_pre_bf region is reused for Y_bf)
  char* w = (char*)d_ws;
  bf16_t* xpre_bf = (bf16_t*)(w + 0);           // 12,582,912 B (dead after GEMM1)
  bf16_t* Y_bf    = (bf16_t*)(w + 0);           // 18,874,368 B (after GEMM2)
  bf16_t* H_bf    = (bf16_t*)(w + 18874368);    // 25,165,824 B
  bf16_t* W1t     = (bf16_t*)(w + 44040192);    //    262,144 B
  bf16_t* W2t     = (bf16_t*)(w + 44302336);    //    786,432 B
  float*  lq      = (float*) (w + 45088768);    //  2,359,296 B
  bf16_t* lq_bf   = (bf16_t*)(w + 47448064);    //  1,179,648 B
  float*  dlq     = (float*) (w + 48627712);    //  2,359,296 B
  float*  sq2     = (float*) (w + 50987008);    //     49,152 B
  float*  S       = (float*) (w + 51036160);    //      3,072 B
  float*  rsq     = (float*) (w + 51039232);    //      3,072 B
  int*    idx     = (int*)   (w + 51042304);    //     24,576 B   (~51 MB total)

  // output layout (floats, concatenated): x | edges | logprobs
  float* x_out = (float*)d_out;                 // 2*16*768*768
  float* edges = x_out + 2L * 16 * N * N;       // 2 * 16*768*8
  float* lp    = edges + 2L * BP * N * K;       // 16*768*8

  dim3 blk(256);

  // one-time bf16 preparation
  convert_bf16_kernel<<<(unsigned)((M * F / 4 + 255) / 256), blk, 0, stream>>>(
      x_pre, xpre_bf, M * F / 4);
  transpose_bf16_kernel<<<dim3(F / 32, Hd / 32), blk, 0, stream>>>(W1, W1t, F, Hd);
  transpose_bf16_kernel<<<dim3(Hd / 32, N / 32), blk, 0, stream>>>(W2, W2t, Hd, N);

  // MLP: two async-staged bf16 WMMA GEMMs (compute-bound core)
  gemm_bf16_wmma_kernel<<<dim3(M / 128, Hd / 128), blk, 0, stream>>>(
      xpre_bf, W1t, b1, nullptr, H_bf, (int)M, Hd, F, /*leaky=*/1);
  gemm_bf16_wmma_kernel<<<dim3(M / 128, N / 128), blk, 0, stream>>>(
      H_bf, W2t, b2, x_out, nullptr, (int)M, N, Hd, /*leaky=*/0);

  // xs = x[0]: bf16 copy for the K=12288 WMMA GEMM; row sums of squares in f32
  convert_bf16_kernel<<<(unsigned)(((long)BP * NN / 4 + 255) / 256), blk, 0, stream>>>(
      x_out, Y_bf, (long)BP * NN / 4);
  rowsq_kernel<<<BP * N, blk, 0, stream>>>(x_out, sq2, N);
  sumS_kernel<<<(N + 255) / 256, blk, 0, stream>>>(sq2, S, N, BP);

  // lq = temp * (S_i + S_j - 2 * sum_b xs_b xs_b^T)
  pairsq_wmma_kernel<<<dim3(N / 128, N / 128), blk, 0, stream>>>(
      Y_bf, S, temperature, lq, N, N, BP, NN);

  // dlq = rsq_i + rsq_j - 2 * lq lq^T
  rowsq_kernel<<<N, blk, 0, stream>>>(lq, rsq, N);
  convert_bf16_kernel<<<(unsigned)((NN / 4 + 255) / 256), blk, 0, stream>>>(
      lq, lq_bf, NN / 4);
  pairsq_wmma_kernel<<<dim3(N / 128, N / 128), blk, 0, stream>>>(
      lq_bf, rsq, nullptr, dlq, N, N, 1, 0);

  // top-8 nearest rows, edges, logprobs
  topk_kernel<<<N, blk, 0, stream>>>(dlq, idx, N, K);
  edges_kernel<<<(BP * N * K + 255) / 256, blk, 0, stream>>>(idx, edges, N, K, BP);
  logprobs_kernel<<<BP * N, blk, 0, stream>>>(x_out, sq2, idx, temperature, lp, N, N, K);
}
